// XXMULT_9096740733618
// MI455X (gfx1250) — matrix operand order repaired and checked by reading, not compiled, not run on hardware
//
#include <hip/hip_runtime.h>
#include <hip/hip_bf16.h>

typedef __attribute__((ext_vector_type(16))) _Float16 v16h;
typedef __attribute__((ext_vector_type(8)))  _Float16 v8h;
typedef __attribute__((ext_vector_type(8)))  float    v8f;

#define T_IN   32
#define EXTRA  3
#define DEPTH  9
#define NNODES 511
#define ROWB   (T_IN + EXTRA)   // 35
#define NS     5
#define PP     2560
#define BBATCH 512
#define KX     (PP / NS)        // 512
#define OUTD   8

#define K0P     96              // leaf K pad (67 -> 96)
#define K1P     224             // inner K: 96 feats + 64 even + 64 odd
#define FSTRIDE 96

// workspace byte offsets
#define WS_WEMB  0              // 64*32 halfs
#define WS_W0P   4096           // 64*96 halfs
#define WS_W1P   16384          // 64*224 halfs
#define WS_FC1   45056          // 64*64 halfs
#define WS_FC2P  53248          // 16*64 halfs (rows 8..15 zero)
#define WS_H0G   55296          // 2560*64 f32
#define WS_MEANT 710656         // 64*512 halfs (transposed mean_h)

// ---------- fragment helpers (16-bit WMMA operand layouts, CDNA5 ISA) ----------
// A 16x32 f16: lane<16 holds K {0..7,16..23}; lane>=16 shifted by +8.
__device__ __forceinline__ v16h afrag_h(const _Float16* row, int kb) {
    v8h lo = *(const v8h*)(row + kb);
    v8h hi = *(const v8h*)(row + kb + 16);
    v16h r;
#pragma unroll
    for (int i = 0; i < 8; ++i) { r[i] = lo[i]; r[i + 8] = hi[i]; }
    return r;
}
__device__ __forceinline__ v16h afrag_f32(const float* row, int kb) {
    v16h r;
#pragma unroll
    for (int i = 0; i < 8; ++i) {
        r[i]     = (_Float16)row[kb + i];
        r[i + 8] = (_Float16)row[kb + 16 + i];
    }
    return r;
}
// B 32x16 f16: lane<16 holds K {0..7,16..23} of column n=lane; lane>=16 shifted by +8.
// Weights stored (N,K) row-major -> two runs of 8 contiguous halfs.
__device__ __forceinline__ v16h bfrag(const _Float16* wrow, int kb) {
    v8h lo = *(const v8h*)(wrow + (kb >> 1));
    v8h hi = *(const v8h*)(wrow + (kb >> 1) + 16);
    v16h r;
#pragma unroll
    for (int i = 0; i < 8; ++i) { r[i] = lo[i]; r[i + 8] = hi[i]; }
    return r;
}
__device__ __forceinline__ v8f wmma16(v16h a, v16h b, v8f c) {
    return __builtin_amdgcn_wmma_f32_16x16x32_f16(false, a, false, b, (short)0, c, false, false);
}

// ---------- weight prep: f32 -> f16, K-padded / segment-aligned ----------
__global__ void prep_weights(const float* __restrict__ W_emb, const float* __restrict__ W0,
                             const float* __restrict__ W1, const float* __restrict__ Wfc1,
                             const float* __restrict__ Wfc2, char* __restrict__ ws) {
    _Float16* wemb = (_Float16*)(ws + WS_WEMB);
    _Float16* w0p  = (_Float16*)(ws + WS_W0P);
    _Float16* w1p  = (_Float16*)(ws + WS_W1P);
    _Float16* fc1  = (_Float16*)(ws + WS_FC1);
    _Float16* fc2p = (_Float16*)(ws + WS_FC2P);
    const int tid = blockIdx.x * blockDim.x + threadIdx.x;
    const int nth = gridDim.x * blockDim.x;
    for (int i = tid; i < 64 * 32; i += nth) wemb[i] = (_Float16)W_emb[i];
    for (int i = tid; i < 64 * K0P; i += nth) {
        int n = i / K0P, k = i % K0P;
        w0p[i] = (_Float16)((k < 67) ? W0[n * 67 + k] : 0.f);
    }
    for (int i = tid; i < 64 * K1P; i += nth) {
        int n = i / K1P, k = i % K1P;
        float v = 0.f;
        if (k < 67)                 v = W1[n * 195 + k];              // feats segment
        else if (k >= 96 && k < 160) v = W1[n * 195 + 67 + (k - 96)]; // h_even
        else if (k >= 160)           v = W1[n * 195 + 131 + (k - 160)]; // h_odd
        w1p[i] = (_Float16)v;
    }
    for (int i = tid; i < 64 * 64; i += nth) fc1[i] = (_Float16)Wfc1[i];
    for (int i = tid; i < 16 * 64; i += nth) {
        int n = i / 64, k = i % 64;
        fc2p[i] = (_Float16)((n < OUTD) ? Wfc2[n * 64 + k] : 0.f);
    }
}

// ---------- fused tree kernel: one workgroup per p ----------
__launch_bounds__(256)
__global__ void tree_kernel(const float* __restrict__ x0, const float* __restrict__ b_emb,
                            const float* __restrict__ b0, const float* __restrict__ b1,
                            char* __restrict__ ws) {
    __shared__ __attribute__((aligned(32))) _Float16 sFeats[8][16 * FSTRIDE]; // 24 KB
    __shared__ __attribute__((aligned(32))) _Float16 sHA[256 * 64];           // 32 KB
    __shared__ __attribute__((aligned(32))) _Float16 sHB[128 * 64];           // 16 KB

    const _Float16* wemb = (const _Float16*)(ws + WS_WEMB);
    const _Float16* w0p  = (const _Float16*)(ws + WS_W0P);
    const _Float16* w1p  = (const _Float16*)(ws + WS_W1P);
    float* h0g = (float*)(ws + WS_H0G);

    const int p    = blockIdx.x;
    const int wave = threadIdx.x >> 5;
    const int lane = threadIdx.x & 31;
    const int ln   = lane & 15;
    const int hi16 = lane >> 4;
    const int basea = hi16 ? 8 : 0;   // A K base
    const int baseb = hi16 ? 16 : 0;  // B K base
    const int mbase = hi16 ? 8 : 0;   // C/D row base

    const float* x0p = x0 + (size_t)p * NNODES * ROWB;

    for (int lev = DEPTH - 1; lev >= 0; --lev) {
        const int lo    = (1 << lev) - 1;
        const int n     = 1 << lev;
        const int tiles = (n + 15) >> 4;
        const bool leaf = (lev == DEPTH - 1);
        _Float16* hCur        = ((DEPTH - 1 - lev) & 1) ? sHB : sHA;
        const _Float16* hPrev = ((DEPTH - 1 - lev) & 1) ? sHA : sHB;
        const float* bias = leaf ? b0 : b1;

        for (int tile = wave; tile < tiles; tile += 8) {
            const int tbase = tile << 4;
            int iLoc = tbase + ln;
            if (iLoc >= n) iLoc = n - 1;               // clamp (small top levels)
            const float* xrow = x0p + (size_t)(lo + iLoc) * ROWB;
            _Float16* ft = sFeats[wave];

            // ---- embedding GEMM: feats cols 0..63 = x0[:, :32] @ W_emb^T + b_emb ----
            v16h ax = afrag_f32(xrow, basea);
#pragma unroll
            for (int nt = 0; nt < 4; ++nt) {
                v16h b = bfrag(wemb + (nt * 16 + ln) * T_IN, baseb);
                v8f c = {};
                c = wmma16(ax, b, c);
                float be = b_emb[nt * 16 + ln];
#pragma unroll
                for (int r = 0; r < 8; ++r)
                    ft[(r + mbase) * FSTRIDE + nt * 16 + ln] = (_Float16)(c[r] + be);
            }
            // ---- extras (cols 64..66) + zero pad (67..95) ----
#pragma unroll
            for (int j = 0; j < 16; ++j) {
                const int row = j;
                const int col = 64 + lane;
                int rl = tbase + row;
                if (rl >= n) rl = n - 1;
                float v = (col < 67) ? x0p[(size_t)(lo + rl) * ROWB + T_IN + (col - 64)] : 0.f;
                ft[row * FSTRIDE + col] = (_Float16)v;
            }
            asm volatile("s_wait_dscnt 0x0" ::: "memory");

            // ---- node GEMM ----
#pragma unroll
            for (int nt = 0; nt < 4; ++nt) {
                v8f acc = {};
                if (leaf) {
                    const _Float16* wrow = w0p + (nt * 16 + ln) * K0P;
#pragma unroll
                    for (int ks = 0; ks < 3; ++ks)
                        acc = wmma16(afrag_h(ft + ln * FSTRIDE + ks * 32, basea),
                                     bfrag(wrow + ks * 32, baseb), acc);
                } else {
                    const _Float16* wrow = w1p + (nt * 16 + ln) * K1P;
#pragma unroll
                    for (int ks = 0; ks < 3; ++ks)
                        acc = wmma16(afrag_h(ft + ln * FSTRIDE + ks * 32, basea),
                                     bfrag(wrow + ks * 32, baseb), acc);
                    const _Float16* he = hPrev + (size_t)(2 * iLoc) * 64; // even child row
                    const _Float16* ho = he + 64;                         // odd child row
#pragma unroll
                    for (int ks = 0; ks < 2; ++ks) {
                        acc = wmma16(afrag_h(he + ks * 32, basea),
                                     bfrag(wrow + 96 + ks * 32, baseb), acc);
                        acc = wmma16(afrag_h(ho + ks * 32, basea),
                                     bfrag(wrow + 160 + ks * 32, baseb), acc);
                    }
                }
                float bi = bias[nt * 16 + ln];
#pragma unroll
                for (int r = 0; r < 8; ++r) {
                    int m = r + mbase;
                    if (tbase + m < n) {
                        float v = acc[r] + bi;
                        hCur[(tbase + m) * 64 + nt * 16 + ln] = (_Float16)(v > 0.f ? v : 0.f);
                    }
                }
            }
        }
        __syncthreads();
    }
    // after lev==0, result row lives in sHA (even level count from leaf)
    if (threadIdx.x < 64)
        h0g[(size_t)p * 64 + threadIdx.x] = (float)sHA[threadIdx.x];
}

// ---------- mean over NS + transpose into B-friendly (64, 512) f16 ----------
__global__ void mean_t_kernel(char* __restrict__ ws) {
    const float* h0g = (const float*)(ws + WS_H0G);
    _Float16* meanT  = (_Float16*)(ws + WS_MEANT);
    int idx = blockIdx.x * blockDim.x + threadIdx.x;
    if (idx >= 64 * KX) return;
    int nn = idx & 63;
    int k  = idx >> 6;
    float s = 0.f;
#pragma unroll
    for (int j = 0; j < NS; ++j) s += h0g[(size_t)(k * NS + j) * 64 + nn];
    meanT[nn * KX + k] = (_Float16)(s * (1.f / NS));
}

// ---------- head: weighted = x @ mean_h; fc1+relu; fc2 ----------
__launch_bounds__(128)
__global__ void head_kernel(const float* __restrict__ x, const float* __restrict__ bfc1,
                            const float* __restrict__ bfc2, const char* __restrict__ ws,
                            float* __restrict__ out) {
    __shared__ __attribute__((aligned(32))) _Float16 sW[16 * 64];
    __shared__ __attribute__((aligned(32))) _Float16 sH[16 * 64];
    const _Float16* meanT = (const _Float16*)(ws + WS_MEANT);
    const _Float16* fc1   = (const _Float16*)(ws + WS_FC1);
    const _Float16* fc2p  = (const _Float16*)(ws + WS_FC2P);

    const int mt   = blockIdx.x;
    const int wave = threadIdx.x >> 5;
    const int lane = threadIdx.x & 31;
    const int ln   = lane & 15;
    const int hi16 = lane >> 4;
    const int basea = hi16 ? 8 : 0, baseb = hi16 ? 16 : 0, mbase = hi16 ? 8 : 0;

    // stage 1: weighted tile (16 x 64), K = 512, wave w owns N-tile w
    {
        const float* xrow = x + (size_t)(mt * 16 + ln) * KX;
        v8f acc = {};
        for (int ks = 0; ks < 16; ++ks) {
            v16h a = afrag_f32(xrow + ks * 32, basea);
            v16h b = bfrag(meanT + (wave * 16 + ln) * KX + ks * 32, baseb);
            acc = wmma16(a, b, acc);
        }
#pragma unroll
        for (int r = 0; r < 8; ++r)
            sW[(r + mbase) * 64 + wave * 16 + ln] = (_Float16)acc[r];
    }
    __syncthreads();
    // stage 2: hid = relu(weighted @ Wfc1^T + bfc1)
    {
        v8f acc = {};
#pragma unroll
        for (int ks = 0; ks < 2; ++ks)
            acc = wmma16(afrag_h(sW + ln * 64 + ks * 32, basea),
                         bfrag(fc1 + (wave * 16 + ln) * 64 + ks * 32, baseb), acc);
        float bi = bfc1[wave * 16 + ln];
#pragma unroll
        for (int r = 0; r < 8; ++r) {
            float v = acc[r] + bi;
            sH[(r + mbase) * 64 + wave * 16 + ln] = (_Float16)(v > 0.f ? v : 0.f);
        }
    }
    __syncthreads();
    // stage 3: out = hid @ Wfc2^T + bfc2 (N padded to 16, write 8)
    if (wave == 0) {
        v8f acc = {};
#pragma unroll
        for (int ks = 0; ks < 2; ++ks)
            acc = wmma16(afrag_h(sH + ln * 64 + ks * 32, basea),
                         bfrag(fc2p + ln * 64 + ks * 32, baseb), acc);
        if (ln < OUTD) {
            float bi = bfc2[ln];
#pragma unroll
            for (int r = 0; r < 8; ++r) {
                int m = r + mbase;
                out[(size_t)(mt * 16 + m) * OUTD + ln] = acc[r] + bi;
            }
        }
    }
}

extern "C" void kernel_launch(void* const* d_in, const int* in_sizes, int n_in,
                              void* d_out, int out_size, void* d_ws, size_t ws_size,
                              hipStream_t stream) {
    (void)in_sizes; (void)n_in; (void)out_size; (void)ws_size;
    const float* x     = (const float*)d_in[0];
    const float* x0    = (const float*)d_in[1];
    const float* W_emb = (const float*)d_in[2];
    const float* b_emb = (const float*)d_in[3];
    const float* W0    = (const float*)d_in[4];
    const float* b0    = (const float*)d_in[5];
    const float* W1    = (const float*)d_in[6];
    const float* b1    = (const float*)d_in[7];
    const float* Wfc1  = (const float*)d_in[8];
    const float* bfc1  = (const float*)d_in[9];
    const float* Wfc2  = (const float*)d_in[10];
    const float* bfc2  = (const float*)d_in[11];
    char* ws   = (char*)d_ws;
    float* out = (float*)d_out;

    prep_weights<<<32, 256, 0, stream>>>(W_emb, W0, W1, Wfc1, Wfc2, ws);
    tree_kernel<<<PP, 256, 0, stream>>>(x0, b_emb, b0, b1, ws);
    mean_t_kernel<<<(64 * KX + 255) / 256, 256, 0, stream>>>(ws);
    head_kernel<<<BBATCH / 16, 128, 0, stream>>>(x, bfc1, bfc2, ws, out);
}
